// NanochatMiniAttention_34368328303155
// MI455X (gfx1250) — matrix operand order repaired and checked by reading, not compiled, not run on hardware
//
#include <hip/hip_runtime.h>
#include <hip/hip_bf16.h>

#define B_ 2
#define S_ 2048
#define E_ 1024
#define NH_ 16
#define NKV_ 4
#define HD_ 64
#define GATE_CH_ 12
#define QKVN_ 1536   // NH*HD + 2*NKV*HD

typedef __attribute__((ext_vector_type(16))) __bf16 v16bf;
typedef __attribute__((ext_vector_type(8)))  __bf16 v8bf;
typedef __attribute__((ext_vector_type(8)))  float  v8f;
typedef __attribute__((ext_vector_type(4)))  unsigned v4u;
typedef __attribute__((ext_vector_type(8)))  int      v8i;
typedef __attribute__((ext_vector_type(4)))  int      v4i;

__device__ __forceinline__ unsigned short f2bf(float f) {
    __bf16 h = (__bf16)f;
    return __builtin_bit_cast(unsigned short, h);
}
__device__ __forceinline__ v8bf ld8(const unsigned short* p) {
    return *reinterpret_cast<const v8bf*>(p);
}
__device__ __forceinline__ v16bf cat16(v8bf lo, v8bf hi) {
    v16bf r;
#pragma unroll
    for (int i = 0; i < 8; ++i) { r[i] = lo[i]; r[i + 8] = hi[i]; }
    return r;
}
__device__ __forceinline__ v8f wmma_bf16(v16bf a, v16bf b, v8f c) {
    return __builtin_amdgcn_wmma_f32_16x16x32_bf16(false, a, false, b, (short)0, c, false, false);
}
__device__ __forceinline__ float wsum32(float v) {
#pragma unroll
    for (int m = 1; m < 32; m <<= 1) v += __shfl_xor(v, m, 32);
    return v;
}
// 16-lane-half reductions: xor masks < 16 never cross the half boundary,
// matching the WMMA C/D layout (rows r / r+8 live in lane halves).
__device__ __forceinline__ float hmax16(float v) {
#pragma unroll
    for (int m = 1; m < 16; m <<= 1) v = fmaxf(v, __shfl_xor(v, m, 32));
    return v;
}
__device__ __forceinline__ float hsum16(float v) {
#pragma unroll
    for (int m = 1; m < 16; m <<= 1) v += __shfl_xor(v, m, 32);
    return v;
}

// ---------------------------------------------------------------- TDM helper
// Issue a Tensor Data Mover 2D tile load (global -> LDS), bf16 elements.
// D# packing per CDNA5 ISA ch.8: group0 = {count/type/addr}, group1 = dims.
// tile_d0/tensor_d0 = contiguous (fastest) dim in elements; stride0 in elements.
__device__ __forceinline__ void tdm_load_2d_bf16(unsigned lds_off, const void* gptr,
                                                 unsigned tensor_d0, unsigned tensor_d1,
                                                 unsigned tile_d0, unsigned tile_d1,
                                                 unsigned stride0) {
    unsigned long long ga = (unsigned long long)gptr;
    v4u g0;
    g0.x = 1u;                                           // count=1, user mode
    g0.y = lds_off;                                      // lds_addr (bytes)
    g0.z = (unsigned)(ga & 0xffffffffu);                 // global_addr[31:0]
    g0.w = (unsigned)((ga >> 32) & 0x01ffffffu) | (2u << 30);  // addr[56:32] | type=2
    v8i g1;
    g1[0] = (int)(1u << 16);                             // data_size=1 (2 bytes)
    g1[1] = (int)((tensor_d0 & 0xffffu) << 16);          // tensor_dim0[15:0]
    g1[2] = (int)(((tensor_d0 >> 16) & 0xffffu) | ((tensor_d1 & 0xffffu) << 16));
    g1[3] = (int)(((tensor_d1 >> 16) & 0xffffu) | ((tile_d0 & 0xffffu) << 16));
    g1[4] = (int)(tile_d1 & 0xffffu);                    // tile_dim1 | tile_dim2=0
    g1[5] = (int)stride0;                                // tensor_dim0_stride[31:0]
    g1[6] = 0;
    g1[7] = 0;
    v4i z4 = {0, 0, 0, 0};
#if __clang_major__ >= 23
    v8i z8 = {0, 0, 0, 0, 0, 0, 0, 0};
    __builtin_amdgcn_tensor_load_to_lds(g0, g1, z4, z4, z8, 0);
#else
    __builtin_amdgcn_tensor_load_to_lds(g0, g1, z4, z4, 0);
#endif
}

// ---------------------------------------------------------------- cvt f32->bf16
__global__ void cvt_f32_bf16(const float* __restrict__ src,
                             unsigned short* __restrict__ dst, int n) {
    int i = blockIdx.x * blockDim.x + threadIdx.x;
    if (i < n) dst[i] = f2bf(src[i]);
}

// ---------------------------------------------------------------- GEMM: C = A * W^T
// A: [M][K] bf16 row-major, W: [N][K] bf16 row-major, C: [M][N] f32.
// 4 waves per block; wave = 16(M) x 64(N) strip; K stepped by 32 (bf16 WMMA).
__global__ __launch_bounds__(128) void gemm_bf16nt(const unsigned short* __restrict__ A,
                                                   const unsigned short* __restrict__ W,
                                                   float* __restrict__ C,
                                                   int M, int N, int K) {
    const int lane  = threadIdx.x & 31;
    const int wave  = threadIdx.x >> 5;
    const int mbase = blockIdx.y * 64 + wave * 16;
    const int nb    = blockIdx.x * 64;
    const int l15   = lane & 15;
    const int khalf = lane >> 4;

    v8f acc[4];
#pragma unroll
    for (int t = 0; t < 4; ++t) acc[t] = (v8f){};

    const unsigned short* arow = A + (size_t)(mbase + l15) * K;

    for (int kb = 0; kb < K; kb += 32) {
        // A fragment: lane row = mbase + l15; K = khalf*8 + e (lo), 16+khalf*8+e (hi)
        v16bf a = cat16(ld8(arow + kb + khalf * 8),
                        ld8(arow + kb + 16 + khalf * 8));
        if (kb + 32 < K) __builtin_prefetch(arow + kb + 32, 0, 3);  // global_prefetch_b8
#pragma unroll
        for (int t = 0; t < 4; ++t) {
            // B fragment: B[k][n] = W[n][k]; per lane K = kb + khalf*8 + e (lo), kb+16+khalf*8+e (hi)
            const unsigned short* wrow =
                W + (size_t)(nb + t * 16 + l15) * K + kb + khalf * 8;
            v16bf b = cat16(ld8(wrow), ld8(wrow + 16));
            acc[t] = wmma_bf16(a, b, acc[t]);
        }
    }
#pragma unroll
    for (int t = 0; t < 4; ++t) {
        const int ncol = nb + t * 16 + l15;
#pragma unroll
        for (int r = 0; r < 8; ++r) {
            const int mrow = mbase + r + 8 * khalf;  // C/D layout: lanes 16-31 hold M=r+8
            C[(size_t)mrow * N + ncol] = acc[t][r];
        }
    }
}

// ---------------------------------------------------------------- epilogue
// One wave per (b,s): gate+ve on v, RoPE + RMS-norm*1.2 on q,k; emit bf16
// head-major tensors qh[B][NH][S][HD], kh/vh[B][NKV][S][HD].
__global__ __launch_bounds__(32) void epilogue(const float* __restrict__ qkv,
                                               const float* __restrict__ x,
                                               const float* __restrict__ ve,
                                               const float* __restrict__ cosp,
                                               const float* __restrict__ sinp,
                                               const float* __restrict__ Wgate,
                                               unsigned short* __restrict__ qh,
                                               unsigned short* __restrict__ kh,
                                               unsigned short* __restrict__ vh) {
    const int bs   = blockIdx.x;           // 0 .. B*S-1
    const int b    = bs / S_;
    const int s    = bs % S_;
    const int lane = threadIdx.x;          // lane L holds hd pair (L, L+32)

    const float cv = cosp[s * (HD_ / 2) + lane];
    const float sv = sinp[s * (HD_ / 2) + lane];
    const float eps = 1.1920929e-7f;

    // gate[j] = 3*sigmoid(x[bs,:12] . Wgate[j])
    float g[NKV_];
    const float xg = (lane < GATE_CH_) ? x[(size_t)bs * E_ + lane] : 0.0f;
#pragma unroll
    for (int j = 0; j < NKV_; ++j) {
        float p = (lane < GATE_CH_) ? xg * Wgate[j * GATE_CH_ + lane] : 0.0f;
        float d = wsum32(p);
        g[j] = 3.0f / (1.0f + __expf(-d));
    }

    const float* row0 = qkv + (size_t)bs * QKVN_;
    // Q heads: RoPE + RMS
    for (int h = 0; h < NH_; ++h) {
        const float* r = row0 + h * HD_;
        float x1 = r[lane], x2 = r[lane + 32];
        float r1 = x1 * cv + x2 * sv;
        float r2 = -x1 * sv + x2 * cv;
        float tot = wsum32(r1 * r1 + r2 * r2);
        float scl = rsqrtf(tot * (1.0f / HD_) + eps) * 1.2f;
        unsigned short* o = qh + ((size_t)(b * NH_ + h) * S_ + s) * HD_;
        o[lane]      = f2bf(r1 * scl);
        o[lane + 32] = f2bf(r2 * scl);
    }
    // K heads: RoPE + RMS
    for (int j = 0; j < NKV_; ++j) {
        const float* r = row0 + NH_ * HD_ + j * HD_;
        float x1 = r[lane], x2 = r[lane + 32];
        float r1 = x1 * cv + x2 * sv;
        float r2 = -x1 * sv + x2 * cv;
        float tot = wsum32(r1 * r1 + r2 * r2);
        float scl = rsqrtf(tot * (1.0f / HD_) + eps) * 1.2f;
        unsigned short* o = kh + ((size_t)(b * NKV_ + j) * S_ + s) * HD_;
        o[lane]      = f2bf(r1 * scl);
        o[lane + 32] = f2bf(r2 * scl);
    }
    // V heads: + gate*ve
    for (int j = 0; j < NKV_; ++j) {
        const float* r  = row0 + (NH_ + NKV_) * HD_ + j * HD_;
        const float* vr = ve + (size_t)bs * (NKV_ * HD_) + j * HD_;
        float v1 = r[lane]      + g[j] * vr[lane];
        float v2 = r[lane + 32] + g[j] * vr[lane + 32];
        unsigned short* o = vh + ((size_t)(b * NKV_ + j) * S_ + s) * HD_;
        o[lane]      = f2bf(v1);
        o[lane + 32] = f2bf(v2);
    }
}

// ---------------------------------------------------------------- flash attention
// One wave per (b, h, 16-query tile); 32 keys per step. K/V tiles staged into
// LDS by the Tensor Data Mover (TDM), fragments read back via DS loads.
__global__ __launch_bounds__(32) void attn(const unsigned short* __restrict__ qh,
                                           const unsigned short* __restrict__ kh,
                                           const unsigned short* __restrict__ vh,
                                           unsigned short* __restrict__ yh) {
    const int lane  = threadIdx.x;
    const int qt    = blockIdx.x;          // S/16 tiles
    const int h     = blockIdx.y;          // NH
    const int b     = blockIdx.z;          // B
    const int j     = h / (NH_ / NKV_);    // GQA kv head
    const int qbase = qt * 16;
    const int l15   = lane & 15;
    const int khalf = lane >> 4;

    const unsigned short* Q  = qh + (size_t)(b * NH_  + h) * S_ * HD_;
    const unsigned short* Kh = kh + (size_t)(b * NKV_ + j) * S_ * HD_;
    const unsigned short* Vh = vh + (size_t)(b * NKV_ + j) * S_ * HD_;

    __shared__ __align__(16) unsigned short Kt[32 * HD_];   // 4 KB K tile
    __shared__ __align__(16) unsigned short Vt[32 * HD_];   // 4 KB V tile
    __shared__ __align__(16) unsigned short Pl[16 * 32];    // P round-trip

    // LDS byte offsets for the TDM descriptors (low 32 bits of generic ptr).
    const unsigned kt_off = (unsigned)(size_t)(void*)&Kt[0];
    const unsigned vt_off = (unsigned)(size_t)(void*)&Vt[0];

    // Q A-fragments, two HD chunks of 32
    const unsigned short* qrow = Q + (size_t)(qbase + l15) * HD_;
    const v16bf aq0 = cat16(ld8(qrow + khalf * 8),      ld8(qrow + 16 + khalf * 8));
    const v16bf aq1 = cat16(ld8(qrow + 32 + khalf * 8), ld8(qrow + 48 + khalf * 8));

    v8f o[4];
#pragma unroll
    for (int t = 0; t < 4; ++t) o[t] = (v8f){};
    float m[8], l[8];
#pragma unroll
    for (int r = 0; r < 8; ++r) { m[r] = -1e30f; l[r] = 0.0f; }

    const int qend = qbase + 15;
    for (int kb = 0; kb <= qend; kb += 32) {
        // --- TDM: stage K and V 32x64 bf16 tiles into LDS (per-wave issue) ---
        tdm_load_2d_bf16(kt_off, Kh + (size_t)kb * HD_, HD_, 32, HD_, 32, HD_);
        tdm_load_2d_bf16(vt_off, Vh + (size_t)kb * HD_, HD_, 32, HD_, 32, HD_);
        __builtin_amdgcn_s_wait_tensorcnt((short)0);

        // scores: two 16-key subtiles, each accumulated over HD in 2 WMMAs.
        // B fragment B[k=hd][n=key]: per lane hd = khalf*8+e (lo), 16+khalf*8+e (hi) at row key.
        v8f s[2];
#pragma unroll
        for (int t = 0; t < 2; ++t) {
            const unsigned short* krow = &Kt[(t * 16 + l15) * HD_];
            v16bf bk0 = cat16(ld8(krow + khalf * 8),      ld8(krow + 16 + khalf * 8));
            v16bf bk1 = cat16(ld8(krow + 32 + khalf * 8), ld8(krow + 48 + khalf * 8));
            v8f z = (v8f){};
            z = wmma_bf16(aq0, bk0, z);
            z = wmma_bf16(aq1, bk1, z);
            s[t] = z;
        }
        // scale, causal mask, online softmax (row = r + 8*khalf, col = lane%16)
        float sc[8];
#pragma unroll
        for (int r = 0; r < 8; ++r) {
            const int qrow_i = qbase + r + 8 * khalf;
            float s0 = (kb + l15      <= qrow_i) ? s[0][r] * 0.125f : -1e30f;
            float s1 = (kb + 16 + l15 <= qrow_i) ? s[1][r] * 0.125f : -1e30f;
            float rmax = hmax16(fmaxf(s0, s1));
            float mnew = fmaxf(m[r], rmax);
            sc[r] = __expf(m[r] - mnew);
            float p0 = __expf(s0 - mnew);
            float p1 = __expf(s1 - mnew);
            s[0][r] = p0; s[1][r] = p1;
            l[r] = l[r] * sc[r] + hsum16(p0 + p1);
            m[r] = mnew;
        }
#pragma unroll
        for (int t = 0; t < 4; ++t)
#pragma unroll
            for (int r = 0; r < 8; ++r) o[t][r] *= sc[r];

        // P: C-layout -> LDS -> A-fragment layout
#pragma unroll
        for (int t = 0; t < 2; ++t)
#pragma unroll
            for (int r = 0; r < 8; ++r)
                Pl[(r + 8 * khalf) * 32 + t * 16 + l15] = f2bf(s[t][r]);
        asm volatile("s_wait_dscnt 0" ::: "memory");
        const unsigned short* prow = &Pl[l15 * 32];
        v16bf ap = cat16(ld8(prow + khalf * 16), ld8(prow + khalf * 16 + 8));

        // O += P * V : B[k=key][n=hd], per-lane K = kb + khalf*16 + e (from LDS tile)
#pragma unroll
        for (int t = 0; t < 4; ++t) {
            v16bf bv;
#pragma unroll
            for (int e = 0; e < 16; ++e)
                bv[e] = __builtin_bit_cast(
                    __bf16, Vt[(khalf * 16 + e) * HD_ + t * 16 + l15]);
            o[t] = wmma_bf16(ap, bv, o[t]);
        }
    }

    // normalize and emit y[b][q][h*64+hd] bf16
#pragma unroll
    for (int r = 0; r < 8; ++r) {
        const float inv = 1.0f / l[r];
        const int q = qbase + r + 8 * khalf;
        unsigned short* yrow = yh + (size_t)(b * S_ + q) * (NH_ * HD_) + h * HD_;
#pragma unroll
        for (int t = 0; t < 4; ++t)
            yrow[t * 16 + l15] = f2bf(o[t][r] * inv);
    }
}

// ---------------------------------------------------------------- launch
extern "C" void kernel_launch(void* const* d_in, const int* in_sizes, int n_in,
                              void* d_out, int out_size, void* d_ws, size_t ws_size,
                              hipStream_t stream) {
    const float* x     = (const float*)d_in[0];
    const float* ve    = (const float*)d_in[1];
    const float* cosp  = (const float*)d_in[2];
    const float* sinp  = (const float*)d_in[3];
    // d_in[4] = attn_mask (causal) — implemented analytically in attn kernel
    const float* Wq    = (const float*)d_in[5];
    const float* Wk    = (const float*)d_in[6];
    const float* Wv    = (const float*)d_in[7];
    const float* Wproj = (const float*)d_in[8];
    const float* Wgate = (const float*)d_in[9];
    float* out = (float*)d_out;

    char* ws = (char*)d_ws;
    size_t off = 0;
    unsigned short* xh  = (unsigned short*)(ws + off); off += (size_t)B_ * S_ * E_ * 2;
    unsigned short* wh  = (unsigned short*)(ws + off); off += (size_t)QKVN_ * E_ * 2;
    unsigned short* wph = (unsigned short*)(ws + off); off += (size_t)E_ * E_ * 2;
    float*          qkv = (float*)(ws + off);          off += (size_t)B_ * S_ * QKVN_ * 4;
    unsigned short* qh  = (unsigned short*)(ws + off); off += (size_t)B_ * NH_  * S_ * HD_ * 2;
    unsigned short* kh  = (unsigned short*)(ws + off); off += (size_t)B_ * NKV_ * S_ * HD_ * 2;
    unsigned short* vh  = (unsigned short*)(ws + off); off += (size_t)B_ * NKV_ * S_ * HD_ * 2;
    unsigned short* yh  = (unsigned short*)(ws + off); off += (size_t)B_ * S_ * E_ * 2;

    const int nx = B_ * S_ * E_;
    cvt_f32_bf16<<<(nx + 255) / 256, 256, 0, stream>>>(x, xh, nx);
    cvt_f32_bf16<<<(1024 * 1024 + 255) / 256, 256, 0, stream>>>(Wq, wh, 1024 * 1024);
    cvt_f32_bf16<<<(256 * 1024 + 255) / 256, 256, 0, stream>>>(Wk, wh + 1024 * 1024, 256 * 1024);
    cvt_f32_bf16<<<(256 * 1024 + 255) / 256, 256, 0, stream>>>(Wv, wh + 1280 * 1024, 256 * 1024);
    cvt_f32_bf16<<<(1024 * 1024 + 255) / 256, 256, 0, stream>>>(Wproj, wph, 1024 * 1024);

    dim3 g1(QKVN_ / 64, (B_ * S_) / 64);
    gemm_bf16nt<<<g1, 128, 0, stream>>>(xh, wh, qkv, B_ * S_, QKVN_, E_);

    epilogue<<<B_ * S_, 32, 0, stream>>>(qkv, x, ve, cosp, sinp, Wgate, qh, kh, vh);

    dim3 ga(S_ / 16, NH_, B_);
    attn<<<ga, 32, 0, stream>>>(qh, kh, vh, yh);

    dim3 g2(E_ / 64, (B_ * S_) / 64);
    gemm_bf16nt<<<g2, 128, 0, stream>>>(yh, wph, out, B_ * S_, E_, E_);
}
